// LSHAttention_51041391346303
// MI455X (gfx1250) — matrix operand order repaired and checked by reading, not compiled, not run on hardware
//
#include <hip/hip_runtime.h>
#include <hip/hip_bf16.h>
#include <math.h>

// ---------------------------------------------------------------------------
// LSH attention (Reformer-style) for gfx1250 (MI455X).
// B=16, S=4096, D=64, N_HASHES=4, BUCKET_SIZE=64, n_buckets=64
//
// Pipeline:
//   k1: hash tokens -> bucket ids (incl. hash offset), [B, 4*S]
//   k2: per-batch 256-bin histogram + exclusive scan -> bucket bases
//   k3: stable compaction (counting-sort scatter) -> sorted index array s
//   k4: per-chunk attention with WMMA f16->f32, scatter results un-sorted
//   k5: combine 4 hash rounds via softmax over per-row logsumexp
// ---------------------------------------------------------------------------

typedef __attribute__((ext_vector_type(16))) _Float16 v16h;
typedef __attribute__((ext_vector_type(8)))  _Float16 v8h;
typedef __attribute__((ext_vector_type(8)))  float    v8f;

#define BATCH 16
#define SEQ   4096
#define DIM   64
#define NH    4
#define NBUCK 64            // buckets per hash
#define TOTB  256           // NH * NBUCK  (bucket ids incl. hash offset)
#define NITEM 16384         // NH * SEQ    (items per batch)
#define NCHUNK 256          // NH * NBUCK chunks of 64 sorted items
#define NEG_MAX (-3.402823466e+38f)
#define SELF_VAL (-50000.0f)

// ---------------------------------------------------------------------------
// Kernel 1: hashing.  rotated[b,h,t,i] = sum_f qk[b,t,f] * rot[f,h,i]
// bucket = argmax over [rot ; -rot] (first max wins), + h*64.
// ---------------------------------------------------------------------------
__global__ __launch_bounds__(128)
void lsh_hash_kernel(const float* __restrict__ qk,
                     const float* __restrict__ rot,   // [64][4][32] flat
                     int* __restrict__ bucketArr) {   // [B][NITEM]
    __shared__ float rotLDS[64 * 128];
    const int tid = threadIdx.x;
    for (int k = tid; k < 64 * 128; k += 128) rotLDS[k] = rot[k];
    __syncthreads();

    const int token = blockIdx.x * 128 + tid;          // b*S + t
    const int b = token >> 12;
    const int t = token & (SEQ - 1);

    float q[64];
    const float* qrow = qk + (size_t)token * DIM;
#pragma unroll
    for (int f = 0; f < 64; f += 4) {
        float4 x = *(const float4*)(qrow + f);
        q[f] = x.x; q[f + 1] = x.y; q[f + 2] = x.z; q[f + 3] = x.w;
    }

    for (int h = 0; h < NH; ++h) {
        float d[32];
#pragma unroll
        for (int ii = 0; ii < 32; ++ii) {
            float acc = 0.0f;
#pragma unroll
            for (int f = 0; f < 64; ++f)
                acc = fmaf(q[f], rotLDS[f * 128 + h * 32 + ii], acc);
            d[ii] = acc;
        }
        float bestv = d[0];
        int besti = 0;
#pragma unroll
        for (int idx = 1; idx < 32; ++idx)
            if (d[idx] > bestv) { bestv = d[idx]; besti = idx; }
#pragma unroll
        for (int idx = 32; idx < 64; ++idx) {
            float nv = -d[idx - 32];
            if (nv > bestv) { bestv = nv; besti = idx; }
        }
        bucketArr[(b << 14) + (h << 12) + t] = besti + h * NBUCK;
    }
}

// ---------------------------------------------------------------------------
// Kernel 2: per-batch histogram over 256 buckets + exclusive prefix scan.
// ---------------------------------------------------------------------------
__global__ __launch_bounds__(256)
void lsh_hist_scan_kernel(const int* __restrict__ bucketArr,
                          int* __restrict__ baseWS) {   // [B][256]
    __shared__ int h[TOTB];
    const int tid = threadIdx.x;
    const int b = blockIdx.x;
    h[tid] = 0;
    __syncthreads();
    for (int it = 0; it < NITEM / TOTB; ++it)
        atomicAdd(&h[bucketArr[b * NITEM + it * TOTB + tid]], 1);
    __syncthreads();
    const int my = h[tid];
    // Hillis-Steele inclusive scan
    for (int d = 1; d < TOTB; d <<= 1) {
        int cur = h[tid];
        int add = (tid >= d) ? h[tid - d] : 0;
        __syncthreads();
        h[tid] = cur + add;
        __syncthreads();
    }
    baseWS[b * TOTB + tid] = h[tid] - my;   // exclusive
}

// ---------------------------------------------------------------------------
// Kernel 3: stable compaction.  One block per (batch, bucket): scan the 16384
// items in index order, place matching items at base + stable rank.
// ---------------------------------------------------------------------------
__global__ __launch_bounds__(256)
void lsh_compact_kernel(const int* __restrict__ bucketArr,
                        const int* __restrict__ baseWS,
                        int* __restrict__ sArr) {       // [B][NITEM]
    const int tid = threadIdx.x;
    const int b = blockIdx.x >> 8;
    const int w = blockIdx.x & 255;
    const int basePos = baseWS[b * TOTB + w];
    const int lane = tid & 31;
    const int wv = tid >> 5;

    __shared__ int waveCnt[8];
    __shared__ int runningSh;
    if (tid == 0) runningSh = 0;
    __syncthreads();

    for (int it = 0; it < NITEM / 256; ++it) {
        const int i = it * 256 + tid;
        const int pred = (bucketArr[b * NITEM + i] == w) ? 1 : 0;
        unsigned mm = (unsigned)__ballot(pred);       // wave32: low 32 bits
        int lp = __popc(mm & ((1u << lane) - 1u));
        if (lane == 0) waveCnt[wv] = __popc(mm);
        __syncthreads();
        int off = 0, tot = 0;
        for (int q2 = 0; q2 < 8; ++q2) {
            int c_ = waveCnt[q2];
            if (q2 < wv) off += c_;
            tot += c_;
        }
        const int running = runningSh;
        if (pred)
            sArr[b * NITEM + basePos + running + off + lp] = i;
        __syncthreads();
        if (tid == 0) runningSh = running + tot;
        __syncthreads();
    }
}

// ---------------------------------------------------------------------------
// WMMA A-operand loader: lane half 'sub' holds
// K = {sub*16 + 0..15} of a 32-wide K slice (matches B operand).
// ---------------------------------------------------------------------------
__device__ __forceinline__ v16h load_wmma_a(const _Float16* base, int sub) {
    const v8h lo = *(const v8h*)(base + sub * 16);
    const v8h hi = *(const v8h*)(base + sub * 16 + 8);
    v16h out;
#pragma unroll
    for (int i = 0; i < 8; ++i) { out[i] = lo[i]; out[i + 8] = hi[i]; }
    return out;
}

// ---------------------------------------------------------------------------
// Kernel 4: chunked attention.  One 256-thread block (8 waves) per
// (batch, chunk).  Stages gathered Q (pre-scaled by D^-0.5), normalized K
// (current + previous chunk), V^T into LDS as f16, runs two WMMA GEMMs with
// masked softmax (f32) between them, scatters un-sorted output + logits.
// ---------------------------------------------------------------------------
__global__ __launch_bounds__(256)
void lsh_attn_kernel(const float* __restrict__ qk,
                     const float* __restrict__ v,
                     const int* __restrict__ bucketArr,
                     const int* __restrict__ sArr,
                     float* __restrict__ oAll,      // [B][NH][S][D]
                     float* __restrict__ logitsWS)  // [B][NH][S]
{
    __shared__ _Float16 bqLDS[64 * 64];    // Q * 0.125, rows of chunk c
    __shared__ _Float16 bkLDS[128 * 64];   // normalized K, chunk c then c-1
    __shared__ _Float16 bvT[64 * 128];     // V^T: [dim e][kv row]
    __shared__ _Float16 pLDS[64 * 128];    // exp(dots - rowmax)
    __shared__ int   kT[128];              // token position of kv rows
    __shared__ int   kBuck[128];           // bucket id of kv rows
    __shared__ float rowredMax[64 * 2];
    __shared__ float rowredSum[64 * 2];
    __shared__ float rowSumF[64];

    const int tid = threadIdx.x;
    const int batch = blockIdx.x >> 8;
    const int c = blockIdx.x & 255;
    const int cprev = (c + NCHUNK - 1) & (NCHUNK - 1);

    // ---- stage: gather rows, normalize K, transpose V ----
    const int g16 = tid >> 4;   // 0..15: row group
    const int l16 = tid & 15;   // 0..15: 4-float segment
    for (int p = 0; p < 8; ++p) {
        const int rr = p * 16 + g16;                       // 0..127
        const int jg = (rr < 64) ? (c * 64 + rr) : (cprev * 64 + (rr - 64));
        const int i = sArr[batch * NITEM + jg];
        const int t = i & (SEQ - 1);
        const size_t rowoff = ((size_t)(batch * SEQ + t)) * DIM;

        const float4 qv = *(const float4*)(qk + rowoff + l16 * 4);
        float ss = qv.x * qv.x + qv.y * qv.y + qv.z * qv.z + qv.w * qv.w;
#pragma unroll
        for (int msk = 1; msk < 16; msk <<= 1) ss += __shfl_xor(ss, msk, 32);
        const float inv = 1.0f / fmaxf(sqrtf(ss), 1e-12f);

        _Float16* bkrow = &bkLDS[rr * 64 + l16 * 4];
        bkrow[0] = (_Float16)(qv.x * inv);
        bkrow[1] = (_Float16)(qv.y * inv);
        bkrow[2] = (_Float16)(qv.z * inv);
        bkrow[3] = (_Float16)(qv.w * inv);
        if (rr < 64) {
            _Float16* bqrow = &bqLDS[rr * 64 + l16 * 4];
            bqrow[0] = (_Float16)(qv.x * 0.125f);   // fold D^-0.5
            bqrow[1] = (_Float16)(qv.y * 0.125f);
            bqrow[2] = (_Float16)(qv.z * 0.125f);
            bqrow[3] = (_Float16)(qv.w * 0.125f);
        }
        const float4 vv = *(const float4*)(v + rowoff + l16 * 4);
        bvT[(l16 * 4 + 0) * 128 + rr] = (_Float16)vv.x;
        bvT[(l16 * 4 + 1) * 128 + rr] = (_Float16)vv.y;
        bvT[(l16 * 4 + 2) * 128 + rr] = (_Float16)vv.z;
        bvT[(l16 * 4 + 3) * 128 + rr] = (_Float16)vv.w;
        if (l16 == 0) {
            kT[rr] = t;
            kBuck[rr] = bucketArr[batch * NITEM + i];
        }
    }
    __syncthreads();

    // ---- GEMM1: dots = Q(64x64) * K^T(64x128) via 16x16x32 f16 WMMA ----
    const int w = tid >> 5;
    const int lane = tid & 31;
    const int sub = lane >> 4;      // half of wave
    const int r = lane & 15;
    const int rt = w >> 1;          // row tile 0..3
    const int cbase = (w & 1) * 4;  // col tiles [cbase, cbase+4)

    float dreg[4][8];
#pragma unroll
    for (int ct4 = 0; ct4 < 4; ++ct4) {
        const int ct = cbase + ct4;
        v8f acc = {};
#pragma unroll
        for (int kb = 0; kb < 2; ++kb) {
            const v16h a = load_wmma_a(&bqLDS[(rt * 16 + r) * 64 + kb * 32], sub);
            const v16h bm = *(const v16h*)&bkLDS[(ct * 16 + r) * 64 + kb * 32 + sub * 16];
            acc = __builtin_amdgcn_wmma_f32_16x16x32_f16(
                false, a, false, bm, (short)0, acc, false, false);
        }
#pragma unroll
        for (int g = 0; g < 8; ++g) {
            const int m = rt * 16 + g + sub * 8;
            const int col = ct * 16 + r;
            float val = acc[g];
            if (kT[m] == kT[col])       val = SELF_VAL;   // self-attn mask
            if (kBuck[m] != kBuck[col]) val = NEG_MAX;    // cross-bucket mask
            dreg[ct4][g] = val;
        }
    }

    // ---- masked softmax across the 128 kv columns ----
#pragma unroll
    for (int g = 0; g < 8; ++g) {
        float rm = fmaxf(fmaxf(dreg[0][g], dreg[1][g]),
                         fmaxf(dreg[2][g], dreg[3][g]));
#pragma unroll
        for (int msk = 1; msk < 16; msk <<= 1) rm = fmaxf(rm, __shfl_xor(rm, msk, 32));
        if (r == 0) rowredMax[(rt * 16 + g + sub * 8) * 2 + (w & 1)] = rm;
    }
    __syncthreads();
#pragma unroll
    for (int g = 0; g < 8; ++g) {
        const int m = rt * 16 + g + sub * 8;
        const float rmax = fmaxf(rowredMax[m * 2], rowredMax[m * 2 + 1]);
        float s = 0.0f;
#pragma unroll
        for (int ct4 = 0; ct4 < 4; ++ct4) {
            const float pv = __expf(dreg[ct4][g] - rmax);
            s += pv;
            pLDS[m * 128 + (cbase + ct4) * 16 + r] = (_Float16)pv;
        }
#pragma unroll
        for (int msk = 1; msk < 16; msk <<= 1) s += __shfl_xor(s, msk, 32);
        if (r == 0) rowredSum[m * 2 + (w & 1)] = s;
    }
    __syncthreads();
    if (tid < 64) {
        const int m = tid;
        const float rmax = fmaxf(rowredMax[2 * m], rowredMax[2 * m + 1]);
        const float rsum = rowredSum[2 * m] + rowredSum[2 * m + 1];
        rowSumF[m] = rsum;
        const float lse = rmax + __logf(rsum);
        const int i = sArr[batch * NITEM + c * 64 + m];
        const int h = i >> 12;
        const int t = i & (SEQ - 1);
        logitsWS[((batch * NH + h) << 12) + t] = lse;
    }
    __syncthreads();

    // ---- GEMM2: bo = P(64x128) * V(128x64), normalized, scatter ----
#pragma unroll
    for (int tsel = 0; tsel < 2; ++tsel) {
        const int tt = w + tsel * 8;
        const int rt2 = tt >> 2;
        const int et = tt & 3;
        v8f acc = {};
#pragma unroll
        for (int kb = 0; kb < 4; ++kb) {
            const v16h a = load_wmma_a(&pLDS[(rt2 * 16 + r) * 128 + kb * 32], sub);
            const v16h bm = *(const v16h*)&bvT[(et * 16 + r) * 128 + kb * 32 + sub * 16];
            acc = __builtin_amdgcn_wmma_f32_16x16x32_f16(
                false, a, false, bm, (short)0, acc, false, false);
        }
#pragma unroll
        for (int g = 0; g < 8; ++g) {
            const int m = rt2 * 16 + g + sub * 8;
            const int e = et * 16 + r;
            const float val = acc[g] / rowSumF[m];
            const int i = sArr[batch * NITEM + c * 64 + m];
            const int h = i >> 12;
            const int t = i & (SEQ - 1);
            oAll[((size_t)(batch * NH + h) * SEQ + t) * DIM + e] = val;
        }
    }
}

// ---------------------------------------------------------------------------
// Kernel 5: combine the NH hash rounds with softmax over per-round lse.
// ---------------------------------------------------------------------------
__global__ __launch_bounds__(256)
void lsh_combine_kernel(const float* __restrict__ oAll,
                        const float* __restrict__ logitsWS,
                        float* __restrict__ out) {
    const int idx = blockIdx.x * 256 + threadIdx.x;   // [B*S*D)
    const int e = idx & 63;
    const int token = idx >> 6;                       // b*S + t
    const int t = token & (SEQ - 1);
    const int b = token >> 12;

    float l[NH];
#pragma unroll
    for (int h = 0; h < NH; ++h) l[h] = logitsWS[((b * NH + h) << 12) + t];
    float m = fmaxf(fmaxf(l[0], l[1]), fmaxf(l[2], l[3]));
    float p[NH], den = 0.0f;
#pragma unroll
    for (int h = 0; h < NH; ++h) { p[h] = __expf(l[h] - m); den += p[h]; }
    float acc = 0.0f;
#pragma unroll
    for (int h = 0; h < NH; ++h)
        acc += p[h] * oAll[((size_t)(b * NH + h) * SEQ + t) * DIM + e];
    out[(size_t)token * DIM + e] = acc / den;
}

// ---------------------------------------------------------------------------
extern "C" void kernel_launch(void* const* d_in, const int* in_sizes, int n_in,
                              void* d_out, int out_size, void* d_ws, size_t ws_size,
                              hipStream_t stream) {
    const float* qk  = (const float*)d_in[0];
    const float* v   = (const float*)d_in[1];
    const float* rot = (const float*)d_in[2];
    float* out = (float*)d_out;

    char* ws = (char*)d_ws;
    float* oAll = (float*)ws;           ws += (size_t)BATCH * NH * SEQ * DIM * sizeof(float);
    float* logitsWS = (float*)ws;       ws += (size_t)BATCH * NH * SEQ * sizeof(float);
    int* bucketArr = (int*)ws;          ws += (size_t)BATCH * NITEM * sizeof(int);
    int* sArr = (int*)ws;               ws += (size_t)BATCH * NITEM * sizeof(int);
    int* baseWS = (int*)ws;             ws += (size_t)BATCH * TOTB * sizeof(int);

    lsh_hash_kernel<<<(BATCH * SEQ) / 128, 128, 0, stream>>>(qk, rot, bucketArr);
    lsh_hist_scan_kernel<<<BATCH, 256, 0, stream>>>(bucketArr, baseWS);
    lsh_compact_kernel<<<BATCH * TOTB, 256, 0, stream>>>(bucketArr, baseWS, sArr);
    lsh_attn_kernel<<<BATCH * NCHUNK, 256, 0, stream>>>(qk, v, bucketArr, sArr,
                                                        oAll, logitsWS);
    lsh_combine_kernel<<<(BATCH * SEQ * DIM) / 256, 256, 0, stream>>>(oAll, logitsWS, out);
}
